// GCN_Geo_78314433675917
// MI455X (gfx1250) — matrix operand order repaired and checked by reading, not compiled, not run on hardware
//
#include <hip/hip_runtime.h>

typedef __attribute__((ext_vector_type(16))) __bf16 v16bf;
typedef __attribute__((ext_vector_type(8)))  float  v8f;

#define D_IN 32
#define D_E  16
#define H1   100
#define H2   20
#define F1   50
#define SLOTS 17      // 16 edge features + 1 bias slot
#define NT1   7       // ceil(100/16) N-tiles, layer 1
#define NT2   2       // ceil(20/16)  N-tiles, layer 2
#define KS2   4       // per-slot K padded to 128 = 4 x 32, layer 2

// ---- workspace byte offsets (all 32B aligned) ----
#define OFF_AGG1 0ull                  // 20000*100 f32 = 8,000,000
#define OFF_H1   8000000ull            // 20000*100 f32 = 8,000,000
#define OFF_AGG2 16000000ull           // 20000*20  f32 = 1,600,000
#define OFF_G    17600000ull           // 500*20    f32 = 40,000
#define OFF_W1PF 17640000ull           // 17*7*512 bf16 = 121,856
#define OFF_W2PF 17761856ull           // 17*4*2*512 bf16 = 139,264
#define OFF_R1PF 17901120ull           // 7*512 bf16 = 7,168
#define OFF_R2PF 17908288ull           // 8*512 bf16 = 8,192

__device__ __forceinline__ int kmapA(int j, int half) {
  // 16-bit A fragment: lanes 0-15 get K {0..15}, lanes 16-31 get K {16..31} (matches B packing)
  return half * 16 + j;
}

// ---------------- utility ----------------
__global__ void zerof_kernel(float* __restrict__ p, int n) {
  int i = blockIdx.x * blockDim.x + threadIdx.x;
  if (i < n) p[i] = 0.f;
}

// ---------------- weight repack (fragment-order bf16) ----------------
__global__ void pack_w1_kernel(const float* __restrict__ W, const float* __restrict__ b,
                               __bf16* __restrict__ dst) {
  int idx = blockIdx.x * blockDim.x + threadIdx.x;
  const int total = SLOTS * NT1 * 32 * 16;
  if (idx >= total) return;
  int j = idx & 15;
  int fl = idx >> 4;
  int lane = fl & 31;
  int t = fl >> 5;
  int nt = t % NT1;
  int slot = t / NT1;
  int k = ((lane >> 4) * 16) + j;        // input dim i, 0..31 (B map: k = half*16 + j)
  int n = nt * 16 + (lane & 15);
  float v = 0.f;
  if (n < H1) v = (slot < D_E) ? W[slot * (D_IN * H1) + k * H1 + n] : b[k * H1 + n];
  dst[idx] = (__bf16)v;
}

__global__ void pack_w2_kernel(const float* __restrict__ W, const float* __restrict__ b,
                               __bf16* __restrict__ dst) {
  int idx = blockIdx.x * blockDim.x + threadIdx.x;
  const int total = SLOTS * KS2 * NT2 * 32 * 16;
  if (idx >= total) return;
  int j = idx & 15;
  int fl = idx >> 4;
  int lane = fl & 31;
  int nt = (fl >> 5) & 1;
  int ks = (fl >> 6) & 3;
  int slot = fl >> 8;
  int k = ks * 32 + ((lane >> 4) * 16) + j;   // 0..127 (input dim within slot)
  int n = nt * 16 + (lane & 15);
  float v = 0.f;
  if (k < H1 && n < H2) v = (slot < D_E) ? W[slot * (H1 * H2) + k * H2 + n] : b[k * H2 + n];
  dst[idx] = (__bf16)v;
}

__global__ void pack_r1_kernel(const float* __restrict__ root1, __bf16* __restrict__ dst) {
  int idx = blockIdx.x * blockDim.x + threadIdx.x;
  const int total = NT1 * 32 * 16;
  if (idx >= total) return;
  int j = idx & 15;
  int fl = idx >> 4;
  int lane = fl & 31;
  int nt = fl >> 5;
  int k = ((lane >> 4) * 16) + j;
  int n = nt * 16 + (lane & 15);
  float v = (n < H1) ? root1[k * H1 + n] : 0.f;
  dst[idx] = (__bf16)v;
}

__global__ void pack_r2_kernel(const float* __restrict__ root2, __bf16* __restrict__ dst) {
  int idx = blockIdx.x * blockDim.x + threadIdx.x;
  const int total = KS2 * NT2 * 32 * 16;
  if (idx >= total) return;
  int j = idx & 15;
  int fl = idx >> 4;
  int lane = fl & 31;
  int nt = (fl >> 5) & 1;
  int ks = fl >> 6;
  int k = ks * 32 + ((lane >> 4) * 16) + j;
  int n = nt * 16 + (lane & 15);
  float v = (k < H1 && n < H2) ? root2[k * H2 + n] : 0.f;
  dst[idx] = (__bf16)v;
}

// ---------------- layer 1: per-edge GEMM [128e x 544] @ [544 x 112] ----------------
__global__ __launch_bounds__(256) void edge1_kernel(
    const float* __restrict__ x, const float* __restrict__ ea,
    const int* __restrict__ ei, const v16bf* __restrict__ Bf,
    float* __restrict__ agg1, int E) {
  __shared__ float xs[128 * 33];
  __shared__ float ee[128 * 18];
  int tid = threadIdx.x;
  int e0 = blockIdx.x * 128;
  for (int idx = tid; idx < 128 * D_IN; idx += 256) {
    int m = idx >> 5, i = idx & 31;
    int e = e0 + m;
    float v = 0.f;
    if (e < E) v = x[ei[e] * D_IN + i];
    xs[m * 33 + i] = v;
  }
  for (int idx = tid; idx < 128 * SLOTS; idx += 256) {
    int m = idx / SLOTS, k = idx % SLOTS;
    int e = e0 + m;
    float v = 0.f;
    if (e < E) v = (k < D_E) ? ea[e * D_E + k] : 1.0f;  // slot 16 carries nn-bias
    ee[m * 18 + k] = v;
  }
  __syncthreads();
  int wave = tid >> 5, lane = tid & 31;
  int half = lane >> 4, lr = lane & 15;
  int mA = wave * 16 + lr;
  float xv[16];
#pragma unroll
  for (int j = 0; j < 16; ++j) xv[j] = xs[mA * 33 + kmapA(j, half)];
  v8f acc[NT1] = {};
  for (int slot = 0; slot < SLOTS; ++slot) {
    // batch all 7 B fragments for this K-slot: 14x b128 loads -> 1 wait -> 7 wmmas
    v16bf b[NT1];
#pragma unroll
    for (int nt = 0; nt < NT1; ++nt) b[nt] = Bf[(slot * NT1 + nt) * 32 + lane];
    if (slot + 1 < SLOTS)
      __builtin_prefetch(&Bf[((slot + 1) * NT1) * 32 + lane], 0, 1);
    float s = ee[mA * 18 + slot];
    v16bf a;
#pragma unroll
    for (int j = 0; j < 16; ++j) a[j] = (__bf16)(s * xv[j]);
#pragma unroll
    for (int nt = 0; nt < NT1; ++nt)
      acc[nt] = __builtin_amdgcn_wmma_f32_16x16x32_bf16(false, a, false, b[nt], (short)0,
                                                        acc[nt], false, false);
  }
  // hoist destination row offsets once
  int dofs[8];
#pragma unroll
  for (int r = 0; r < 8; ++r) {
    int e = e0 + wave * 16 + half * 8 + r;
    dofs[r] = (e < E) ? ei[E + e] * H1 : -1;
  }
#pragma unroll
  for (int nt = 0; nt < NT1; ++nt) {
    int n = nt * 16 + lr;
    if (n < H1) {
#pragma unroll
      for (int r = 0; r < 8; ++r)
        if (dofs[r] >= 0) atomicAdd(&agg1[dofs[r] + n], acc[nt][r]);
    }
  }
}

// ---------------- node 1: h1 = relu(agg1 + x@root1 + bias1) ----------------
__global__ __launch_bounds__(256) void node1_kernel(
    const float* __restrict__ x, const float* __restrict__ agg1,
    const v16bf* __restrict__ Bf, const float* __restrict__ bias1,
    float* __restrict__ h1, int N) {
  int tid = threadIdx.x;
  int wave = tid >> 5, lane = tid & 31, half = lane >> 4, lr = lane & 15;
  int n0 = blockIdx.x * 128;
  int na = n0 + wave * 16 + lr;
  v16bf a;
#pragma unroll
  for (int j = 0; j < 16; ++j) {
    float v = (na < N) ? x[na * D_IN + kmapA(j, half)] : 0.f;
    a[j] = (__bf16)v;
  }
  v16bf b[NT1];
#pragma unroll
  for (int nt = 0; nt < NT1; ++nt) b[nt] = Bf[nt * 32 + lane];
#pragma unroll
  for (int nt = 0; nt < NT1; ++nt) {
    v8f c = {};
    c = __builtin_amdgcn_wmma_f32_16x16x32_bf16(false, a, false, b[nt], (short)0, c, false, false);
    int n = nt * 16 + lr;
    if (n < H1) {
#pragma unroll
      for (int r = 0; r < 8; ++r) {
        int node = n0 + wave * 16 + half * 8 + r;
        if (node < N) {
          float v = c[r] + agg1[node * H1 + n] + bias1[n];
          h1[node * H1 + n] = fmaxf(v, 0.f);
        }
      }
    }
  }
}

// ---------------- layer 2: per-edge GEMM [128e x 2176] @ [2176 x 32] ----------------
__global__ __launch_bounds__(256) void edge2_kernel(
    const float* __restrict__ h1, const float* __restrict__ ea,
    const int* __restrict__ ei, const v16bf* __restrict__ Bf,
    float* __restrict__ agg2, int E) {
  __shared__ float hs[128 * 104];
  __shared__ float ee[128 * 18];
  int tid = threadIdx.x;
  int e0 = blockIdx.x * 128;
  for (int idx = tid; idx < 128 * H1; idx += 256) {
    int m = idx / H1, i = idx % H1;
    int e = e0 + m;
    float v = 0.f;
    if (e < E) v = h1[ei[e] * H1 + i];
    hs[m * 104 + i] = v;
  }
  for (int idx = tid; idx < 128 * SLOTS; idx += 256) {
    int m = idx / SLOTS, k = idx % SLOTS;
    int e = e0 + m;
    float v = 0.f;
    if (e < E) v = (k < D_E) ? ea[e * D_E + k] : 1.0f;
    ee[m * 18 + k] = v;
  }
  __syncthreads();
  int wave = tid >> 5, lane = tid & 31;
  int half = lane >> 4, lr = lane & 15;
  int mA = wave * 18;  // row base for ee
  int mH = (wave * 16 + lr);
  mA = mH * 18;
  v8f acc[NT2] = {};
  for (int ks = 0; ks < KS2; ++ks) {
    // hoist this K-subtile of the h1 row into registers (reused by all 17 slots)
    float hv[16];
#pragma unroll
    for (int j = 0; j < 16; ++j) {
      int k = ks * 32 + kmapA(j, half);
      hv[j] = (k < H1) ? hs[mH * 104 + k] : 0.f;
    }
    // slots 0..15 in chunks of 4: 16x b128 loads -> 1 wait -> 8 wmmas
    for (int sb = 0; sb < 16; sb += 4) {
      v16bf b[8];
#pragma unroll
      for (int u = 0; u < 4; ++u)
#pragma unroll
        for (int nt = 0; nt < NT2; ++nt)
          b[u * 2 + nt] = Bf[(((sb + u) * KS2 + ks) * NT2 + nt) * 32 + lane];
      if (sb + 4 < 16)
        __builtin_prefetch(&Bf[(((sb + 4) * KS2 + ks) * NT2) * 32 + lane], 0, 1);
#pragma unroll
      for (int u = 0; u < 4; ++u) {
        float s = ee[mA + sb + u];
        v16bf a;
#pragma unroll
        for (int j = 0; j < 16; ++j) a[j] = (__bf16)(s * hv[j]);
#pragma unroll
        for (int nt = 0; nt < NT2; ++nt)
          acc[nt] = __builtin_amdgcn_wmma_f32_16x16x32_bf16(false, a, false, b[u * 2 + nt],
                                                            (short)0, acc[nt], false, false);
      }
    }
    // bias slot (slot 16)
    {
      v16bf b0 = Bf[((16 * KS2 + ks) * NT2 + 0) * 32 + lane];
      v16bf b1 = Bf[((16 * KS2 + ks) * NT2 + 1) * 32 + lane];
      float s = ee[mA + 16];
      v16bf a;
#pragma unroll
      for (int j = 0; j < 16; ++j) a[j] = (__bf16)(s * hv[j]);
      acc[0] = __builtin_amdgcn_wmma_f32_16x16x32_bf16(false, a, false, b0, (short)0, acc[0], false, false);
      acc[1] = __builtin_amdgcn_wmma_f32_16x16x32_bf16(false, a, false, b1, (short)0, acc[1], false, false);
    }
  }
  int dofs[8];
#pragma unroll
  for (int r = 0; r < 8; ++r) {
    int e = e0 + wave * 16 + half * 8 + r;
    dofs[r] = (e < E) ? ei[E + e] * H2 : -1;
  }
#pragma unroll
  for (int nt = 0; nt < NT2; ++nt) {
    int n = nt * 16 + lr;
    if (n < H2) {
#pragma unroll
      for (int r = 0; r < 8; ++r)
        if (dofs[r] >= 0) atomicAdd(&agg2[dofs[r] + n], acc[nt][r]);
    }
  }
}

// ------- node 2: h2 = relu(agg2 + h1@root2 + bias2), pooled into g via atomics -------
__global__ __launch_bounds__(256) void node2_kernel(
    const float* __restrict__ h1, const float* __restrict__ agg2,
    const v16bf* __restrict__ Bf, const float* __restrict__ bias2,
    const int* __restrict__ batch, float* __restrict__ g, int N) {
  int tid = threadIdx.x;
  int wave = tid >> 5, lane = tid & 31, half = lane >> 4, lr = lane & 15;
  int n0 = blockIdx.x * 128;
  int na = n0 + wave * 16 + lr;
  v16bf b[KS2 * NT2];
#pragma unroll
  for (int q = 0; q < KS2 * NT2; ++q) b[q] = Bf[q * 32 + lane];
  v8f acc[NT2] = {};
#pragma unroll
  for (int ks = 0; ks < KS2; ++ks) {
    v16bf a;
#pragma unroll
    for (int j = 0; j < 16; ++j) {
      int k = ks * 32 + kmapA(j, half);
      float v = (na < N && k < H1) ? h1[na * H1 + k] : 0.f;
      a[j] = (__bf16)v;
    }
#pragma unroll
    for (int nt = 0; nt < NT2; ++nt)
      acc[nt] = __builtin_amdgcn_wmma_f32_16x16x32_bf16(false, a, false, b[ks * NT2 + nt],
                                                        (short)0, acc[nt], false, false);
  }
  int gofs[8];
#pragma unroll
  for (int r = 0; r < 8; ++r) {
    int node = n0 + wave * 16 + half * 8 + r;
    gofs[r] = (node < N) ? batch[node] * H2 : -1;
  }
#pragma unroll
  for (int nt = 0; nt < NT2; ++nt) {
    int n = nt * 16 + lr;
    if (n < H2) {
#pragma unroll
      for (int r = 0; r < 8; ++r) {
        int node = n0 + wave * 16 + half * 8 + r;
        if (gofs[r] >= 0) {
          float v = acc[nt][r] + agg2[node * H2 + n] + bias2[n];
          v = fmaxf(v, 0.f);
          atomicAdd(&g[gofs[r] + n], v);
        }
      }
    }
  }
}

// ---------------- FCN head ----------------
__global__ void head_kernel(const float* __restrict__ g, const float* __restrict__ l1W,
                            const float* __restrict__ l1b, const float* __restrict__ l2W,
                            const float* __restrict__ l2b, float* __restrict__ out, int G) {
  int t = blockIdx.x * blockDim.x + threadIdx.x;
  if (t >= G) return;
  float gr[H2];
#pragma unroll
  for (int k = 0; k < H2; ++k) gr[k] = g[t * H2 + k];
  float o = 0.f;
  for (int f = 0; f < F1; ++f) {
    float h = l1b[f];
#pragma unroll
    for (int k = 0; k < H2; ++k) h += gr[k] * l1W[k * F1 + f];
    o += fmaxf(h, 0.f) * l2W[f];
  }
  out[t] = o + l2b[0];
}

extern "C" void kernel_launch(void* const* d_in, const int* in_sizes, int n_in,
                              void* d_out, int out_size, void* d_ws, size_t ws_size,
                              hipStream_t stream) {
  const float* x     = (const float*)d_in[0];
  const float* ea    = (const float*)d_in[1];
  const int*   ei    = (const int*)d_in[2];
  const int*   batch = (const int*)d_in[3];
  const float* nn1W  = (const float*)d_in[4];
  const float* nn1b  = (const float*)d_in[5];
  const float* root1 = (const float*)d_in[6];
  const float* bias1 = (const float*)d_in[7];
  const float* nn2W  = (const float*)d_in[8];
  const float* nn2b  = (const float*)d_in[9];
  const float* root2 = (const float*)d_in[10];
  const float* bias2 = (const float*)d_in[11];
  const float* l1W   = (const float*)d_in[12];
  const float* l1b   = (const float*)d_in[13];
  const float* l2W   = (const float*)d_in[14];
  const float* l2b   = (const float*)d_in[15];
  float* out = (float*)d_out;

  int N = in_sizes[0] / D_IN;   // 20000
  int E = in_sizes[1] / D_E;    // 40000
  int G = out_size;             // 500

  char* ws = (char*)d_ws;
  float*  agg1 = (float*)(ws + OFF_AGG1);
  float*  h1   = (float*)(ws + OFF_H1);
  float*  agg2 = (float*)(ws + OFF_AGG2);
  float*  g    = (float*)(ws + OFF_G);
  __bf16* w1pf = (__bf16*)(ws + OFF_W1PF);
  __bf16* w2pf = (__bf16*)(ws + OFF_W2PF);
  __bf16* r1pf = (__bf16*)(ws + OFF_R1PF);
  __bf16* r2pf = (__bf16*)(ws + OFF_R2PF);

  // zero accumulators (agg1; agg2+g are contiguous)
  {
    int n1 = N * H1;
    zerof_kernel<<<(n1 + 255) / 256, 256, 0, stream>>>(agg1, n1);
    int n2 = N * H2 + G * H2;
    zerof_kernel<<<(n2 + 255) / 256, 256, 0, stream>>>(agg2, n2);
  }
  // repack weights into fragment-order bf16
  pack_w1_kernel<<<(SLOTS * NT1 * 512 + 255) / 256, 256, 0, stream>>>(nn1W, nn1b, w1pf);
  pack_w2_kernel<<<(SLOTS * KS2 * NT2 * 512 + 255) / 256, 256, 0, stream>>>(nn2W, nn2b, w2pf);
  pack_r1_kernel<<<(NT1 * 512 + 255) / 256, 256, 0, stream>>>(root1, r1pf);
  pack_r2_kernel<<<(KS2 * NT2 * 512 + 255) / 256, 256, 0, stream>>>(root2, r2pf);

  int eb = (E + 127) / 128;
  int nb = (N + 127) / 128;

  edge1_kernel<<<eb, 256, 0, stream>>>(x, ea, ei, (const v16bf*)w1pf, agg1, E);
  node1_kernel<<<nb, 256, 0, stream>>>(x, agg1, (const v16bf*)r1pf, bias1, h1, N);
  edge2_kernel<<<eb, 256, 0, stream>>>(h1, ea, ei, (const v16bf*)w2pf, agg2, E);
  node2_kernel<<<nb, 256, 0, stream>>>(h1, agg2, (const v16bf*)r2pf, bias2, batch, g, N);
  head_kernel<<<(G + 255) / 256, 256, 0, stream>>>(g, l1W, l1b, l2W, l2b, out, G);
}
